// SpectralConv2d_36249523978357
// MI455X (gfx1250) — matrix operand order repaired and checked by reading, not compiled, not run on hardware
//
#include <hip/hip_runtime.h>
#include <hip/hip_bf16.h>

// SpectralConv2d (FNO layer) as 5 WMMA GEMM stages on gfx1250 (wave32).
// B=16, IN_C=OUT_C=64, H=W=256, MX=MY=32.
// Stages 2/4 stage their reusable complex tile in LDS via the Tensor Data
// Mover (TENSOR_LOAD_TO_LDS + s_wait_tensorcnt), then gather WMMA B-fragments
// with paired ds_load_b32 ops.

typedef __attribute__((ext_vector_type(16))) _Float16 v16h;
typedef __attribute__((ext_vector_type(8)))  float    v8f;
typedef unsigned int v4u __attribute__((ext_vector_type(4)));
typedef int          v8i_t __attribute__((ext_vector_type(8)));
typedef int          v4i_t __attribute__((ext_vector_type(4)));

union AFrag { v16h v; _Float16 h[16]; };
union DFrag { v8f  v; float    f[8];  };
union H2    { unsigned u; _Float16 h[2]; };

__device__ __forceinline__ v8f wmma16(v16h a, v16h b, v8f c) {
  return __builtin_amdgcn_wmma_f32_16x16x32_f16(false, a, false, b, (short)0, c, false, false);
}
// A-fragment K index within a 32-wide K step (16-bit A, 16x32 layout)
__device__ __forceinline__ int akf(int lane, int e) { return e + (e & 8) + ((lane & 16) >> 1); }
// B-fragment K index within a 32-wide K step (16-bit B, 32x16 layout)
__device__ __forceinline__ int bkf(int lane, int e) { return e + (e & 8) + ((lane & 16) >> 1); }

#if defined(__has_builtin)
#if __has_builtin(__builtin_amdgcn_tensor_load_to_lds) && __has_builtin(__builtin_amdgcn_s_wait_tensorcnt)
#define HAVE_TDM 1
#endif
#endif

#ifdef HAVE_TDM
// 1D contiguous TDM copy: ndwords dwords from global to LDS offset lds_off.
// D# layout per CDNA5 ISA 8.3/8.4: group0 = {count=1, lds_addr, global_addr,
// type=2}; group1 = {data_size=4B, tensor_dim0=tile_dim0=ndwords, stride0=ndwords}.
__device__ __forceinline__ void tdm_load_1d(unsigned lds_off, const void* gptr, unsigned ndwords) {
  unsigned long long ga = (unsigned long long)(size_t)gptr;
  v4u g0;
  g0.x = 1u;                                               // count=1, user mode
  g0.y = lds_off;                                          // lds_addr (bytes)
  g0.z = (unsigned)(ga & 0xffffffffu);                     // global_addr[31:0]
  g0.w = (unsigned)((ga >> 32) & 0x01ffffffu) | (2u << 30); // addr[56:32] | type=2
  v8i_t g1;
  g1[0] = (int)(2u << 16);                                 // data_size=2 (4 bytes)
  g1[1] = (int)((ndwords & 0xffffu) << 16);                // tensor_dim0[15:0]
  g1[2] = (int)((ndwords >> 16) & 0xffffu);                // tensor_dim0[31:16]
  g1[3] = (int)((ndwords & 0xffffu) << 16);                // tile_dim0
  g1[4] = 0;                                               // tile_dim1/2 unused
  g1[5] = (int)ndwords;                                    // tensor_dim0_stride
  g1[6] = 0;
  g1[7] = 0;
  v4i_t z4 = {0, 0, 0, 0};
#if __clang_major__ >= 23
  v8i_t z8 = {0, 0, 0, 0, 0, 0, 0, 0};
  __builtin_amdgcn_tensor_load_to_lds(g0, g1, z4, z4, z8, 0);
#else
  __builtin_amdgcn_tensor_load_to_lds(g0, g1, z4, z4, 0);
#endif
}
#endif

// Stage a contiguous block of dwords into LDS (TDM if available).
__device__ __forceinline__ void stage_to_lds(_Float16* sh, const _Float16* src, int ndwords) {
#ifdef HAVE_TDM
  if ((threadIdx.x >> 5) == 0) {
    tdm_load_1d((unsigned)(size_t)sh, (const void*)src, (unsigned)ndwords);
    __builtin_amdgcn_s_wait_tensorcnt((short)0);
  }
#else
  const unsigned* s = (const unsigned*)src;
  unsigned* d = (unsigned*)sh;
  for (int i = threadIdx.x; i < ndwords; i += blockDim.x) d[i] = s[i];
#endif
  __syncthreads();
}

// ---------------------------------------------------------------------------
// Init: pack 4 twiddle matrices into WMMA fragment order (f16).
//  sec0: W1pack  B-mat [K=256][N=64]  : forward-W DFT  (cos, -sin)/16
//  sec1: A2pack  A-mat [M=32 ][K=512] : forward-H DFT  (cos, sin)/16, doubled-K
//  sec2: A4pack  A-mat [M=256][K=64]  : inverse-H DFT  (cos, sin)/16, doubled-K
//  sec3: W5pack  B-mat [K=64 ][N=256] : irfft-W        (c_ky*cos, -c_ky*sin)/16
// ---------------------------------------------------------------------------
__global__ void init_twiddles(_Float16* W1p, _Float16* A2p, _Float16* A4p, _Float16* W5p) {
  int idx  = blockIdx.x * blockDim.x + threadIdx.x;   // 0..65535
  int sec  = idx >> 14;
  int j    = idx & 16383;
  int e    = j & 15;
  int lane = (j >> 4) & 31;
  int t    = j >> 9;                                  // tile index 0..31
  const float PI2 = 6.28318530717958647692f;
  const float S = 1.0f / 16.0f;
  if (sec == 0) {                   // t = ks*4 + nt ; ks 0..7, nt 0..3
    int ks = t >> 2, nt = t & 3;
    int k = ks * 32 + bkf(lane, e);        // k == w
    int n = nt * 16 + (lane & 15);
    int ky = n >> 1, c = n & 1;
    float th = PI2 * (float)(ky * k) / 256.0f;
    W1p[j] = (_Float16)(((c == 0) ? __cosf(th) : -__sinf(th)) * S);
  } else if (sec == 1) {            // t = mt*16 + ks ; mt 0..1, ks 0..15
    int ks = t & 15, mt = t >> 4;
    int k = ks * 32 + akf(lane, e);        // doubled K: h = k>>1
    int kx = mt * 16 + (lane & 15);
    int h = k >> 1, odd = k & 1;
    float th = PI2 * (float)(kx * h) / 256.0f;
    A2p[j] = (_Float16)((odd ? __sinf(th) : __cosf(th)) * S);
  } else if (sec == 2) {            // t = hmt*2 + ks ; hmt 0..15, ks 0..1
    int ks = t & 1, hmt = t >> 1;
    int k = ks * 32 + akf(lane, e);        // doubled K: kx = k>>1
    int h = hmt * 16 + (lane & 15);
    int kx = k >> 1, odd = k & 1;
    float th = PI2 * (float)(kx * h) / 256.0f;
    A4p[j] = (_Float16)((odd ? __sinf(th) : __cosf(th)) * S);
  } else {                          // t = ks*16 + ntile ; ks 0..1, ntile 0..15
    int ks = t >> 4, nt = t & 15;
    int k = ks * 32 + bkf(lane, e);        // doubled K: ky = k>>1
    int n = nt * 16 + (lane & 15);         // n == w
    int ky = k >> 1, c = k & 1;
    float cy = (ky == 0) ? 1.0f : 2.0f;    // Hermitian doubling for irfft
    float th = PI2 * (float)(ky * n) / 256.0f;
    W5p[j] = (_Float16)(((c == 0) ? cy * __cosf(th) : -cy * __sinf(th)) * S);
  }
}

// ---------------------------------------------------------------------------
// Stage 1: forward partial DFT along W.
// T1[row=(b*64+c)*256+h][n=2*ky+comp] = sum_w x[row][w] * W1[w][n]
// GEMM: M=262144, K=256, N=64.  Block=256thr(8 waves), wave = 16 rows x 64 cols.
// ---------------------------------------------------------------------------
__global__ void k1_dftW(const float* __restrict__ x, const _Float16* __restrict__ W1p,
                        _Float16* __restrict__ T1) {
  int lane = threadIdx.x & 31, wave = threadIdx.x >> 5;
  int row0 = blockIdx.x * 128 + wave * 16;
  const float* xrow = x + (size_t)(row0 + (lane & 15)) * 256;
  const v16h* bp = (const v16h*)W1p;
  v8f acc0 = {}, acc1 = {}, acc2 = {}, acc3 = {};
  for (int ks = 0; ks < 8; ++ks) {
    AFrag a;
#pragma unroll
    for (int e = 0; e < 16; e += 2) {
      int k = ks * 32 + akf(lane, e);
      float2 p = *(const float2*)(xrow + k);
      a.h[e] = (_Float16)p.x; a.h[e + 1] = (_Float16)p.y;
    }
    acc0 = wmma16(a.v, bp[(ks * 4 + 0) * 32 + lane], acc0);
    acc1 = wmma16(a.v, bp[(ks * 4 + 1) * 32 + lane], acc1);
    acc2 = wmma16(a.v, bp[(ks * 4 + 2) * 32 + lane], acc2);
    acc3 = wmma16(a.v, bp[(ks * 4 + 3) * 32 + lane], acc3);
  }
  int rbase = row0 + ((lane & 16) >> 1);
  int col = lane & 15;
  DFrag d0, d1, d2, d3; d0.v = acc0; d1.v = acc1; d2.v = acc2; d3.v = acc3;
#pragma unroll
  for (int r = 0; r < 8; ++r) {
    size_t ro = (size_t)(rbase + r) * 64;
    T1[ro +  0 + col] = (_Float16)d0.f[r];
    T1[ro + 16 + col] = (_Float16)d1.f[r];
    T1[ro + 32 + col] = (_Float16)d2.f[r];
    T1[ro + 48 + col] = (_Float16)d3.f[r];
  }
}

// ---------------------------------------------------------------------------
// Stage 2: forward partial DFT along H (complex x complex via doubled K=512).
// Per bc: T2[kx][n] = sum_k' A2[kx][k'] * B2[k'][n],  B2 = swizzle(T1[bc]).
// T1 tile (32 KB) staged in LDS via TDM; B-fragments = paired ds_load_b32.
// Grid = 1024 (bc); wave (mt=w>>2, nt=w&3) -> 16x16 tile of 32x64 output.
// ---------------------------------------------------------------------------
__global__ void k2_dftH(const _Float16* __restrict__ T1, const _Float16* __restrict__ A2p,
                        _Float16* __restrict__ T2) {
  __shared__ __align__(16) _Float16 sh[256 * 64];   // 32 KB
  int lane = threadIdx.x & 31, wave = threadIdx.x >> 5;
  int bc = blockIdx.x;
  stage_to_lds(sh, T1 + (size_t)bc * 16384, 8192);

  int mt = wave >> 2, nt = wave & 3;
  const v16h* ap = (const v16h*)A2p;
  int n = nt * 16 + (lane & 15);
  int c = n & 1, ky2 = (n >> 1) * 2;                // element pair base in a row
  v8f acc = {};
  for (int ks = 0; ks < 16; ++ks) {
    v16h a = ap[(mt * 16 + ks) * 32 + lane];
    AFrag b;
#pragma unroll
    for (int e = 0; e < 16; e += 2) {
      int k = ks * 32 + bkf(lane, e);               // even -> h = k>>1
      int h = k >> 1;
      H2 p; p.u = *(const unsigned*)(sh + h * 64 + ky2);  // (re, im)
      // even k': T1[h][n]   -> c ? im : re
      // odd  k': c==0 ? +im : -re
      b.h[e]     = c ? p.h[1] : p.h[0];
      b.h[e + 1] = c ? (_Float16)(-p.h[0]) : p.h[1];
    }
    acc = wmma16(a, b.v, acc);
  }
  DFrag d; d.v = acc;
  int kx0 = mt * 16 + ((lane & 16) >> 1);
#pragma unroll
  for (int r = 0; r < 8; ++r)
    T2[(size_t)bc * 2048 + (kx0 + r) * 64 + n] = (_Float16)d.f[r];
}

// ---------------------------------------------------------------------------
// Stage 3: per-mode complex channel mix. Per (kx,ky): M=16(b), K=128, N=128.
// Grid = 1024 (kx*32+ky); wave = ntile 0..7.
// ---------------------------------------------------------------------------
__global__ void k3_modemix(const _Float16* __restrict__ T2, const float* __restrict__ wr,
                           const float* __restrict__ wi, _Float16* __restrict__ M3) {
  int lane = threadIdx.x & 31, nt = threadIdx.x >> 5;
  int kx = blockIdx.x >> 5, ky = blockIdx.x & 31;
  int n = nt * 16 + (lane & 15);
  int o = n >> 1, cc = n & 1;
  int brow = lane & 15;
  v8f acc = {};
  for (int ks = 0; ks < 4; ++ks) {
    AFrag a;
#pragma unroll
    for (int e = 0; e < 16; e += 2) {               // pair: (Xr, Xi) at (b,i)
      int k = ks * 32 + akf(lane, e);
      int i = k >> 1;
      H2 p; p.u = *(const unsigned*)(T2 + ((size_t)(brow * 64 + i) * 32 + kx) * 64 + 2 * ky);
      a.h[e] = p.h[0]; a.h[e + 1] = p.h[1];
    }
    AFrag b;
#pragma unroll
    for (int e = 0; e < 16; e += 2) {               // pair shares (i,o) weight
      int k = ks * 32 + bkf(lane, e);
      int i = k >> 1;
      size_t widx = ((size_t)(i * 64 + o) * 32 + kx) * 32 + ky;
      float wrv = wr[widx], wiv = wi[widx];
      // B[2i][2o]=Wr, [2i][2o+1]=Wi ; B[2i+1][2o]=-Wi, [2i+1][2o+1]=Wr
      b.h[e]     = (_Float16)(cc ? wiv : wrv);
      b.h[e + 1] = (_Float16)(cc ? wrv : -wiv);
    }
    acc = wmma16(a.v, b.v, acc);
  }
  DFrag d; d.v = acc;
  int b0 = (lane & 16) >> 1;
#pragma unroll
  for (int r = 0; r < 8; ++r) {
    int bb = b0 + r;
    M3[((size_t)(bb * 64 + o) * 32 + kx) * 64 + 2 * ky + cc] = (_Float16)d.f[r];
  }
}

// ---------------------------------------------------------------------------
// Stage 4: inverse partial DFT along H. Per bo: M=256(h), K=64(doubled), N=64.
// M3 tile (4 KB) staged in LDS via TDM; B-fragments = paired ds_load_b32.
// Grid = 8192: bo = idx>>3, hg = idx&7 (32 h rows); wave: mloc=w&1, nt=w>>1.
// ---------------------------------------------------------------------------
__global__ void k4_invH(const _Float16* __restrict__ M3, const _Float16* __restrict__ A4p,
                        _Float16* __restrict__ T3) {
  __shared__ __align__(16) _Float16 shm[32 * 64];   // 4 KB
  int lane = threadIdx.x & 31, wave = threadIdx.x >> 5;
  int bo = blockIdx.x >> 3, hg = blockIdx.x & 7;
  stage_to_lds(shm, M3 + (size_t)bo * 2048, 1024);

  int mloc = wave & 1, nt = wave >> 1;
  int hmt = hg * 2 + mloc;                 // 0..15
  const v16h* ap = (const v16h*)A4p;
  int n = nt * 16 + (lane & 15);
  int c = n & 1, ky2 = (n >> 1) * 2;
  v8f acc = {};
  for (int ks = 0; ks < 2; ++ks) {
    v16h a = ap[(hmt * 2 + ks) * 32 + lane];
    AFrag b;
#pragma unroll
    for (int e = 0; e < 16; e += 2) {
      int k = ks * 32 + bkf(lane, e);               // even -> kx = k>>1
      int kx = k >> 1;
      H2 p; p.u = *(const unsigned*)(shm + kx * 64 + ky2);  // (Mr, Mi)
      // even k': c ? Mi : Mr ; odd k': c ? Mr : -Mi
      b.h[e]     = c ? p.h[1] : p.h[0];
      b.h[e + 1] = c ? p.h[0] : (_Float16)(-p.h[1]);
    }
    acc = wmma16(a, b.v, acc);
  }
  DFrag d; d.v = acc;
  int h0 = hmt * 16 + ((lane & 16) >> 1);
#pragma unroll
  for (int r = 0; r < 8; ++r)
    T3[(size_t)bo * 16384 + (h0 + r) * 64 + n] = (_Float16)d.f[r];
}

// ---------------------------------------------------------------------------
// Stage 5: irfft along W. out[row][w] = sum_{k=0..63} T3[row][k] * W5[k][w].
// GEMM: M=262144, K=64, N=256.  Grid=8192 (32 rows/block); wave: mloc=w&1,
// ng=w>>1 covers 64 cols (4 n-tiles).  Full f32 output written.
// ---------------------------------------------------------------------------
__global__ void k5_invW(const _Float16* __restrict__ T3, const _Float16* __restrict__ W5p,
                        float* __restrict__ out) {
  int lane = threadIdx.x & 31, wave = threadIdx.x >> 5;
  int mloc = wave & 1, ng = wave >> 1;
  int row0 = blockIdx.x * 32 + mloc * 16;
  const _Float16* trow = T3 + (size_t)(row0 + (lane & 15)) * 64;
  const v16h* bp = (const v16h*)W5p;
  v8f acc0 = {}, acc1 = {}, acc2 = {}, acc3 = {};
  for (int ks = 0; ks < 2; ++ks) {
    AFrag a;
#pragma unroll
    for (int e = 0; e < 16; e += 2) {
      int k = ks * 32 + akf(lane, e);
      H2 p; p.u = *(const unsigned*)(trow + k);
      a.h[e] = p.h[0]; a.h[e + 1] = p.h[1];
    }
    acc0 = wmma16(a.v, bp[(ks * 16 + ng * 4 + 0) * 32 + lane], acc0);
    acc1 = wmma16(a.v, bp[(ks * 16 + ng * 4 + 1) * 32 + lane], acc1);
    acc2 = wmma16(a.v, bp[(ks * 16 + ng * 4 + 2) * 32 + lane], acc2);
    acc3 = wmma16(a.v, bp[(ks * 16 + ng * 4 + 3) * 32 + lane], acc3);
  }
  DFrag d0, d1, d2, d3; d0.v = acc0; d1.v = acc1; d2.v = acc2; d3.v = acc3;
  int rbase = row0 + ((lane & 16) >> 1);
  int col = ng * 64 + (lane & 15);
#pragma unroll
  for (int r = 0; r < 8; ++r) {
    size_t ro = (size_t)(rbase + r) * 256 + col;
    out[ro +  0] = d0.f[r];
    out[ro + 16] = d1.f[r];
    out[ro + 32] = d2.f[r];
    out[ro + 48] = d3.f[r];
  }
}

// ---------------------------------------------------------------------------
extern "C" void kernel_launch(void* const* d_in, const int* in_sizes, int n_in,
                              void* d_out, int out_size, void* d_ws, size_t ws_size,
                              hipStream_t stream) {
  const float* x  = (const float*)d_in[0];
  const float* wr = (const float*)d_in[1];
  const float* wi = (const float*)d_in[2];
  float* out = (float*)d_out;

  char* ws = (char*)d_ws;
  _Float16* W1p = (_Float16*)(ws + 0);
  _Float16* A2p = (_Float16*)(ws + 32768);
  _Float16* A4p = (_Float16*)(ws + 65536);
  _Float16* W5p = (_Float16*)(ws + 98304);
  _Float16* T1  = (_Float16*)(ws + 131072);                         // 32 MB
  _Float16* T2  = (_Float16*)(ws + 131072 + 33554432);              //  4 MB
  _Float16* M3  = (_Float16*)(ws + 131072 + 33554432 + 4194304);    //  4 MB
  _Float16* T3  = (_Float16*)(ws + 131072 + 33554432 + 8388608);    // 32 MB

  init_twiddles<<<64, 1024, 0, stream>>>(W1p, A2p, A4p, W5p);
  k1_dftW   <<<2048, 256, 0, stream>>>(x, W1p, T1);
  k2_dftH   <<<1024, 256, 0, stream>>>(T1, A2p, T2);
  k3_modemix<<<1024, 256, 0, stream>>>(T2, wr, wi, M3);
  k4_invH   <<<8192, 256, 0, stream>>>(M3, A4p, T3);
  k5_invW   <<<8192, 256, 0, stream>>>(T3, W5p, out);
}
